// LlamaAttention_77017353552566
// MI455X (gfx1250) — matrix operand order repaired and checked by reading, not compiled, not run on hardware
//
#include <hip/hip_runtime.h>
#include <hip/hip_bf16.h>
#include <math.h>

// ---------------- problem constants (B=1) ----------------
#define S_LEN 2048
#define DMODEL 4096
#define NH 32
#define NKV 8
#define GQA (NH / NKV)   // 4
#define DHEAD 128

typedef __attribute__((ext_vector_type(16))) __bf16 bf16x16;
typedef __attribute__((ext_vector_type(8)))  __bf16 bf16x8;
typedef __attribute__((ext_vector_type(8)))  float  floatx8;
typedef __attribute__((ext_vector_type(4)))  unsigned u32x4;
typedef __attribute__((ext_vector_type(8)))  unsigned u32x8;

// D = A(16x32 bf16) * B(32x16 bf16) + C(16x16 f32)
__device__ __forceinline__ floatx8 wmma_bf16(bf16x16 a, bf16x16 b, floatx8 c) {
    return __builtin_amdgcn_wmma_f32_16x16x32_bf16(
        false, a, false, b, (short)0, c, false, false);
}

// ---- async global->LDS (ASYNCcnt) ----
__device__ __forceinline__ void async_copy_b128(unsigned lds_byte_addr, const void* gaddr) {
    asm volatile("global_load_async_to_lds_b128 %0, %1, off"
                 :: "v"(lds_byte_addr), "v"(gaddr) : "memory");
}
__device__ __forceinline__ void wait_async(int n) {  // n is compile-time 0 or 5
    if (n == 5) asm volatile("s_wait_asynccnt 0x5" ::: "memory");
    else        asm volatile("s_wait_asynccnt 0x0" ::: "memory");
}

// ---- Tensor Data Mover: 2D tile load, 16-bit elements (TENSORcnt) ----
// D# group0/group1 per CDNA5 ISA 8.3/8.4. LDS padding: pad_amount(code)
// DWORDs inserted every pad_interval(code) DWORDs.
__device__ __forceinline__ void tdm_load_2d_b16(
        unsigned lds_addr, const void* gaddr,
        unsigned tile_d0, unsigned tile_d1,
        unsigned d0_stride_elems,
        unsigned pad_interval_code, unsigned pad_amount_code) {
    unsigned long long ga = (unsigned long long)gaddr;
    u32x4 g0;
    g0[0] = 1u;                                            // count=1 (user D#)
    g0[1] = lds_addr;                                      // lds_addr (bytes)
    g0[2] = (unsigned)(ga & 0xFFFFFFFFu);                  // global_addr[31:0]
    g0[3] = (unsigned)((ga >> 32) & 0x01FFFFFFu)           // global_addr[56:32]
          | 0x80000000u;                                   // type=2 ("image")
    u32x8 g1;
    g1[0] = (1u << 16)                                     // data_size = 2 bytes
          | (1u << 20)                                     // pad_enable
          | (pad_interval_code << 22) | (pad_amount_code << 25);
    g1[1] = (tile_d0 & 0xFFFFu) << 16;                     // tensor_dim0[15:0]
    g1[2] = ((tile_d0 >> 16) & 0xFFFFu) | ((tile_d1 & 0xFFFFu) << 16); // dim0 hi | dim1 lo
    g1[3] = ((tile_d1 >> 16) & 0xFFFFu) | ((tile_d0 & 0xFFFFu) << 16); // dim1 hi | tile_dim0
    g1[4] = (tile_d1 & 0xFFFFu);                           // tile_dim1 (tile_dim2=0)
    g1[5] = d0_stride_elems;                               // tensor_dim0_stride lo32
    g1[6] = 0u;
    g1[7] = 0u;
    asm volatile("tensor_load_to_lds %0, %1" :: "s"(g0), "s"(g1) : "memory");
}

__device__ __forceinline__ bf16x16 combine16(bf16x8 a, bf16x8 b) {
    bf16x16 r;
#pragma unroll
    for (int j = 0; j < 8; ++j) { r[j] = a[j]; r[j + 8] = b[j]; }
    return r;
}

// A-fragment (16x32, 16-bit): lane -> row (L&15); one contiguous 32B run at k = 16*hi.
__device__ __forceinline__ bf16x16 load_A_frag(const __bf16* row, int k0, int hi) {
    bf16x8 lo8 = *(const bf16x8*)(row + k0 + 16 * hi);
    bf16x8 hi8 = *(const bf16x8*)(row + k0 + 16 * hi + 8);
    return combine16(lo8, hi8);
}

// ---------------- fp32 -> bf16 convert (row-major copy) ----------------
__global__ void cvt_f32_bf16(const float* __restrict__ src, __bf16* __restrict__ dst, int n) {
    int i = blockIdx.x * blockDim.x + threadIdx.x;
    if (i < n) dst[i] = (__bf16)src[i];
}

// ---------------- fp32 [K][N] -> bf16 [N][K] tiled transpose ----------------
__global__ void cvt_transpose_bf16(const float* __restrict__ src, __bf16* __restrict__ dst,
                                   int K, int N) {
    __shared__ float tile[32][33];
    const int kb = blockIdx.y * 32, nb = blockIdx.x * 32;
    const int tx = threadIdx.x, ty = threadIdx.y;   // (32, 8)
#pragma unroll
    for (int i = 0; i < 32; i += 8)
        tile[ty + i][tx] = src[(size_t)(kb + ty + i) * N + nb + tx];
    __syncthreads();
#pragma unroll
    for (int i = 0; i < 32; i += 8)
        dst[(size_t)(nb + ty + i) * K + kb + tx] = (__bf16)tile[tx][ty + i];
}

// ---- V: fp32 [S][NKV][DH] -> bf16 [NKV][DH][S] (per-head d-major) ----
__global__ void vtrans_bf16(const float* __restrict__ src, __bf16* __restrict__ dst) {
    __shared__ float tile[32][33];
    const int sb = blockIdx.x * 32, db = blockIdx.y * 32, hk = blockIdx.z;
    const int tx = threadIdx.x, ty = threadIdx.y;
#pragma unroll
    for (int i = 0; i < 32; i += 8)
        tile[ty + i][tx] = src[(size_t)(sb + ty + i) * (NKV * DHEAD) + hk * DHEAD + db + tx];
    __syncthreads();
#pragma unroll
    for (int i = 0; i < 32; i += 8)
        dst[(size_t)hk * DHEAD * S_LEN + (size_t)(db + ty + i) * S_LEN + sb + tx]
            = (__bf16)tile[tx][ty + i];
}

// ---------------- GEMM: C[M,N] (f32) = A[M,K] @ Wt[N,K]^T, bf16 -------------
// 256 thr = 8 waves; workgroup tile 64x256; wave tile 32x64; K-step 32.
// Both tiles staged by async b128 copies, double-buffered (5 copies/thread).
#define AS_STRIDE 40
#define BT_STRIDE 40
__global__ void gemm_bf16(const __bf16* __restrict__ A, const __bf16* __restrict__ Wt,
                          float* __restrict__ C, int M, int N, int K) {
    __shared__ __bf16 As[2][64 * AS_STRIDE];
    __shared__ __bf16 Bs[2][256 * BT_STRIDE];
    const int t    = threadIdx.x;
    const int lane = t & 31;
    const int wave = t >> 5;
    const int wm   = wave >> 2;          // 0..1
    const int wn   = wave & 3;           // 0..3
    const int lo   = lane & 15;
    const int hi   = lane >> 4;
    const int mblk = blockIdx.y * 64;
    const int nblk = blockIdx.x * 256;

    // A: thread -> row t>>2, 8-elem chunk (t&3)*8
    const int ar = t >> 2, ac = (t & 3) * 8;
    const __bf16* agp = A + (size_t)(mblk + ar) * K + ac;

    auto issue = [&](int buf, int k0) {
        async_copy_b128((unsigned)(size_t)&As[buf][ar * AS_STRIDE + ac], agp + k0);
#pragma unroll
        for (int i = 0; i < 4; ++i) {
            int g  = t + i * 256;            // 0..1023
            int nr = g >> 2;                 // 0..255
            int kc = (g & 3) * 8;            // 0..24
            async_copy_b128((unsigned)(size_t)&Bs[buf][nr * BT_STRIDE + kc],
                            Wt + (size_t)(nblk + nr) * K + k0 + kc);
        }
    };

    floatx8 acc[2][4];
#pragma unroll
    for (int i = 0; i < 2; ++i)
#pragma unroll
        for (int j = 0; j < 4; ++j) acc[i][j] = (floatx8){};

    const int nk = K / 32;
    issue(0, 0);
    for (int kt = 0; kt < nk; ++kt) {
        const int cur = kt & 1;
        if (kt + 1 < nk) {
            issue(cur ^ 1, (kt + 1) * 32);
            __builtin_prefetch(agp + (size_t)(kt + 2) * 32, 0, 0);
            wait_async(5);                   // current tile resident (in-order)
        } else {
            wait_async(0);
        }
        __syncthreads();                     // all threads' chunks visible

        const __bf16* arow = &As[cur][(wm * 32 + lo) * AS_STRIDE];
        bf16x16 a0 = load_A_frag(arow, 0, hi);
        bf16x16 a1 = load_A_frag(arow + 16 * AS_STRIDE, 0, hi);
#pragma unroll
        for (int j = 0; j < 4; ++j) {
            bf16x16 b = *(const bf16x16*)(&Bs[cur][(wn * 64 + j * 16 + lo) * BT_STRIDE + hi * 16]);
            acc[0][j] = wmma_bf16(a0, b, acc[0][j]);
            acc[1][j] = wmma_bf16(a1, b, acc[1][j]);
        }
        __syncthreads();                     // done reading before overwrite
    }

    const int m0 = mblk + wm * 32;
    const int n0 = nblk + wn * 64;
#pragma unroll
    for (int j = 0; j < 4; ++j)
#pragma unroll
        for (int v = 0; v < 8; ++v) {
            int r = m0 + v + 8 * hi;
            C[(size_t)r * N + n0 + j * 16 + lo]        = acc[0][j][v];
            C[(size_t)(r + 16) * N + n0 + j * 16 + lo] = acc[1][j][v];
        }
}

// ---------------- RoPE (NeoX) + bf16 convert; scale folded into Q ----------
__global__ void rope_cvt(const float* __restrict__ X, __bf16* __restrict__ Y,
                         const int* __restrict__ pos_ids, int nheads, float scale) {
    const int s = blockIdx.x;
    const int h = blockIdx.y;
    const int i = threadIdx.x;               // 0..63
    float pos = (float)pos_ids[s];
    float inv = __powf(10000.0f, -(float)i / 64.0f);
    float ang = pos * inv;
    float cs = __cosf(ang), sn = __sinf(ang);
    const float* xp = X + ((size_t)s * nheads + h) * DHEAD;
    __bf16* yp = Y + ((size_t)s * nheads + h) * DHEAD;
    float x1 = xp[i], x2 = xp[i + 64];
    yp[i]      = (__bf16)((x1 * cs - x2 * sn) * scale);
    yp[i + 64] = (__bf16)((x2 * cs + x1 * sn) * scale);
}

// ---------------- flash attention -------------------------------------------
// Block = 128 thr (4 waves) = one head x 64 q rows. K tile (32x128) and V tile
// (Vt[d][key], from the d-major global layout) staged by the Tensor Data Mover
// with LDS padding giving bank-friendly strides; TENSORcnt + barrier sync.
#define KS_STRIDE 136   // 128 + 4 DWORDs pad per 64 DWORDs (codes 5,3)
#define VT_STRIDE 40    // 32 + 4 DWORDs pad per 16 DWORDs (codes 3,3)
#define PS_STRIDE 40
__global__ void attn_kernel(const __bf16* __restrict__ Qb, const __bf16* __restrict__ Kb,
                            const __bf16* __restrict__ Vt_g, __bf16* __restrict__ Ob) {
    __shared__ __bf16 Ks[32 * KS_STRIDE];
    __shared__ __bf16 Vt[128 * VT_STRIDE];
    __shared__ __bf16 Ps[4 * 16 * PS_STRIDE];
    const int h    = blockIdx.x;             // 0..31
    const int qblk = blockIdx.y;             // 0..31
    const int hk   = h / GQA;
    const int t    = threadIdx.x;            // 0..127
    const int wave = t >> 5;
    const int lane = t & 31;
    const int lo   = lane & 15;
    const int hi   = lane >> 4;
    const int m0   = qblk * 64 + wave * 16;

    bf16x16 qa[4];
    {
        const __bf16* qrow = Qb + ((size_t)(m0 + lo) * NH + h) * DHEAD;
#pragma unroll
        for (int c = 0; c < 4; ++c) qa[c] = load_A_frag(qrow, c * 32, hi);
    }

    float row_max[8], row_sum[8];
    floatx8 o[8];
#pragma unroll
    for (int v = 0; v < 8; ++v) { row_max[v] = -1e30f; row_sum[v] = 0.f; }
#pragma unroll
    for (int c = 0; c < 8; ++c) o[c] = (floatx8){};

    __bf16* ps = Ps + wave * 16 * PS_STRIDE;
    const __bf16* vbase = Vt_g + (size_t)hk * DHEAD * S_LEN;
    const int kend = qblk * 64 + 63;

    for (int kb0 = 0; kb0 <= kend; kb0 += 32) {
        __syncthreads();                     // previous tile reads done
        if (wave == 0) {
            // K tile: 32 rows x 128 elems, row stride NKV*DHEAD
            tdm_load_2d_b16((unsigned)(size_t)Ks,
                            Kb + ((size_t)kb0 * NKV + hk) * DHEAD,
                            128, 32, NKV * DHEAD, /*pad*/ 5, 3);
            // V tile: 128 rows x 32 elems, row stride S
            tdm_load_2d_b16((unsigned)(size_t)Vt,
                            vbase + kb0,
                            32, 128, S_LEN, /*pad*/ 3, 3);
        }
        __builtin_amdgcn_s_wait_tensorcnt(0);
        __syncthreads();                     // tiles visible to all waves

        // ---- scores = Q @ K^T ----
        floatx8 s0 = {}, s1 = {};
#pragma unroll
        for (int c = 0; c < 4; ++c) {
            bf16x16 b0 = *(const bf16x16*)(Ks + lo * KS_STRIDE + c * 32 + hi * 16);
            bf16x16 b1 = *(const bf16x16*)(Ks + (16 + lo) * KS_STRIDE + c * 32 + hi * 16);
            s0 = wmma_bf16(qa[c], b0, s0);
            s1 = wmma_bf16(qa[c], b1, s1);
        }

        // ---- causal mask ----
        if (kb0 + 31 > m0) {
#pragma unroll
            for (int v = 0; v < 8; ++v) {
                int qr = m0 + v + 8 * hi;
                if (kb0 + lo > qr)      s0[v] = -1e30f;
                if (kb0 + 16 + lo > qr) s1[v] = -1e30f;
            }
        }

        // ---- online softmax ----
#pragma unroll
        for (int v = 0; v < 8; ++v) {
            float mx = fmaxf(s0[v], s1[v]);
#pragma unroll
            for (int off = 8; off >= 1; off >>= 1) mx = fmaxf(mx, __shfl_xor(mx, off, 16));
            float mnew = fmaxf(row_max[v], mx);
            float corr = __expf(row_max[v] - mnew);
            row_max[v] = mnew;
            float p0 = __expf(s0[v] - mnew);
            float p1 = __expf(s1[v] - mnew);
            s0[v] = p0; s1[v] = p1;
            float psum = p0 + p1;
#pragma unroll
            for (int off = 8; off >= 1; off >>= 1) psum += __shfl_xor(psum, off, 16);
            row_sum[v] = row_sum[v] * corr + psum;
#pragma unroll
            for (int c = 0; c < 8; ++c) o[c][v] *= corr;
        }

        // ---- P: C-frag -> A-frag through per-wave LDS tile ----
#pragma unroll
        for (int v = 0; v < 8; ++v) {
            ps[(v + 8 * hi) * PS_STRIDE + lo]      = (__bf16)s0[v];
            ps[(v + 8 * hi) * PS_STRIDE + 16 + lo] = (__bf16)s1[v];
        }
        bf16x16 pa = load_A_frag(ps + lo * PS_STRIDE, 0, hi);

        // ---- O += P @ V ----
#pragma unroll
        for (int c = 0; c < 8; ++c) {
            bf16x16 vb = *(const bf16x16*)(Vt + (c * 16 + lo) * VT_STRIDE + hi * 16);
            o[c] = wmma_bf16(pa, vb, o[c]);
        }
    }

    // ---- normalize and store bf16 [S, H, DH] ----
#pragma unroll
    for (int c = 0; c < 8; ++c)
#pragma unroll
        for (int v = 0; v < 8; ++v) {
            float val = o[c][v] / row_sum[v];
            Ob[((size_t)(m0 + v + 8 * hi) * NH + h) * DHEAD + c * 16 + lo] = (__bf16)val;
        }
}

// ---------------- host side ----------------
extern "C" void kernel_launch(void* const* d_in, const int* in_sizes, int n_in,
                              void* d_out, int out_size, void* d_ws, size_t ws_size,
                              hipStream_t stream) {
    const int*   pos = (const int*)d_in[0];
    const float* hs  = (const float*)d_in[1];
    const float* Wq  = (const float*)d_in[2];
    const float* Wk  = (const float*)d_in[3];
    const float* Wv  = (const float*)d_in[4];
    const float* Wo  = (const float*)d_in[5];
    float* out = (float*)d_out;

    const size_t SD  = (size_t)S_LEN * DMODEL;
    const size_t SKV = (size_t)S_LEN * NKV * DHEAD;

    char* ws = (char*)d_ws;
    size_t off = 0;
    auto alloc = [&](size_t bytes) -> void* {
        void* p = ws + off;
        off += (bytes + 255) & ~(size_t)255;
        return p;
    };
    __bf16* hsb  = (__bf16*)alloc(SD * 2);
    __bf16* Wqt  = (__bf16*)alloc((size_t)DMODEL * NH * DHEAD * 2);   // [N][K]
    __bf16* Wkt  = (__bf16*)alloc((size_t)DMODEL * NKV * DHEAD * 2);
    __bf16* Wvt  = (__bf16*)alloc((size_t)DMODEL * NKV * DHEAD * 2);
    __bf16* Wot  = (__bf16*)alloc((size_t)NH * DHEAD * DMODEL * 2);
    float*  Qf   = (float*)alloc(SD * 4);
    float*  Kf   = (float*)alloc(SKV * 4);
    float*  Vf   = (float*)alloc(SKV * 4);
    __bf16* qb   = (__bf16*)alloc(SD * 2);
    __bf16* kb   = (__bf16*)alloc(SKV * 2);
    __bf16* vt   = (__bf16*)alloc(SKV * 2);                           // [NKV][DH][S]
    __bf16* aob  = (__bf16*)alloc(SD * 2);
    (void)ws_size; (void)in_sizes; (void)n_in; (void)out_size;

    const int T = 256;
    cvt_f32_bf16<<<(unsigned)((SD + T - 1) / T), T, 0, stream>>>(hs, hsb, (int)SD);
    // weights -> transposed bf16 [N][K]
    dim3 tb(32, 8);
    cvt_transpose_bf16<<<dim3((NH * DHEAD) / 32, DMODEL / 32), tb, 0, stream>>>(Wq, Wqt, DMODEL, NH * DHEAD);
    cvt_transpose_bf16<<<dim3((NKV * DHEAD) / 32, DMODEL / 32), tb, 0, stream>>>(Wk, Wkt, DMODEL, NKV * DHEAD);
    cvt_transpose_bf16<<<dim3((NKV * DHEAD) / 32, DMODEL / 32), tb, 0, stream>>>(Wv, Wvt, DMODEL, NKV * DHEAD);
    cvt_transpose_bf16<<<dim3(DMODEL / 32, (NH * DHEAD) / 32), tb, 0, stream>>>(Wo, Wot, NH * DHEAD, DMODEL);

    // projections (N tiles of 256, M tiles of 64)
    gemm_bf16<<<dim3((NH * DHEAD) / 256, S_LEN / 64), T, 0, stream>>>(hsb, Wqt, Qf, S_LEN, NH * DHEAD, DMODEL);
    gemm_bf16<<<dim3((NKV * DHEAD) / 256, S_LEN / 64), T, 0, stream>>>(hsb, Wkt, Kf, S_LEN, NKV * DHEAD, DMODEL);
    gemm_bf16<<<dim3((NKV * DHEAD) / 256, S_LEN / 64), T, 0, stream>>>(hsb, Wvt, Vf, S_LEN, NKV * DHEAD, DMODEL);

    // RoPE (scale folded into Q); V -> d-major bf16 for TDM tiles
    const float scale = 1.0f / sqrtf((float)DHEAD);
    rope_cvt<<<dim3(S_LEN, NH), 64, 0, stream>>>(Qf, qb, pos, NH, scale);
    rope_cvt<<<dim3(S_LEN, NKV), 64, 0, stream>>>(Kf, kb, pos, NKV, 1.0f);
    vtrans_bf16<<<dim3(S_LEN / 32, DHEAD / 32, NKV), tb, 0, stream>>>(Vf, vt);

    // attention: 4 waves per block, one head x 64 q rows
    attn_kernel<<<dim3(NH, S_LEN / 64), 128, 0, stream>>>(qb, kb, vt, aob);

    // output projection -> fp32 d_out
    gemm_bf16<<<dim3(DMODEL / 256, S_LEN / 64), T, 0, stream>>>(aob, Wot, out, S_LEN, DMODEL, NH * DHEAD);
}
